// LocalAttentionBlock_27401891349135
// MI455X (gfx1250) — matrix operand order repaired and checked by reading, not compiled, not run on hardware
//
#include <hip/hip_runtime.h>
#include <hip/hip_bf16.h>

// ---------------- problem constants ----------------
#define WIDTH     1024
#define NUM_HEADS 16
#define HEAD_DIM  64
#define WINDOW    256
#define BATCH     2
#define SEQ       2048
#define BT        (BATCH*SEQ)

typedef __attribute__((ext_vector_type(16))) __bf16 v16bf;
typedef __attribute__((ext_vector_type(8)))  __bf16 v8bf;
typedef __attribute__((ext_vector_type(8)))  float  v8f;
typedef unsigned int u32x4 __attribute__((ext_vector_type(4)));
typedef int          i32x4 __attribute__((ext_vector_type(4)));
typedef int          i32x8 __attribute__((ext_vector_type(8)));

#if defined(__has_builtin)
#if __has_builtin(__builtin_amdgcn_tensor_load_to_lds)
#define HAVE_TDM 1
#endif
#endif
#ifndef HAVE_TDM
#define HAVE_TDM 0
#endif

__device__ __forceinline__ v8f wmma_bf16(v16bf a, v16bf b, v8f c) {
  // (neg_a, A, neg_b, B, c_mod, C, reuse_a, reuse_b)
  return __builtin_amdgcn_wmma_f32_16x16x32_bf16(false, a, false, b, (short)0, c, false, false);
}

__device__ __forceinline__ __bf16 f2bf(float f) { return (__bf16)f; }

// A-matrix 16x32 bf16 fragment, contiguous k order (matches B loads):
// lane<16 : M=lane,    K = 0..15
// lane>=16: M=lane-16, K = 16..31
__device__ __forceinline__ v16bf load_a16(const __bf16* p, int hi) {
  v8bf lo = *(const v8bf*)(p + hi * 16);
  v8bf hh = *(const v8bf*)(p + hi * 16 + 8);
  v16bf r;
#pragma unroll
  for (int i = 0; i < 8; ++i) { r[i] = lo[i]; r[i + 8] = hh[i]; }
  return r;
}

#if HAVE_TDM
// TDM: DMA a [rows x rowlen] bf16 strip (row-major, stride == rowlen) from
// global memory to LDS offset 0 of this workgroup. 2D D# per CDNA5 ISA ch.8.
__device__ __forceinline__ void tdm_load_strip(const __bf16* gsrc, unsigned rows,
                                               unsigned rowlen) {
  unsigned long long ga = (unsigned long long)(__UINTPTR_TYPE__)gsrc;
  u32x4 g0;
  g0[0] = 1u;                                              // count=1, user desc
  g0[1] = 0u;                                              // lds_addr = 0
  g0[2] = (unsigned)(ga & 0xFFFFFFFFu);                    // global_addr[31:0]
  g0[3] = (unsigned)((ga >> 32) & 0x01FFFFFFu) | (2u << 30); // addr[56:32], type=2
  i32x8 g1;
  g1[0] = (int)(1u << 16);                // wg_mask=0, data_size=1 (2 bytes)
  g1[1] = (int)(rowlen << 16);            // tensor_dim0[15:0] @ bits 63:48
  g1[2] = (int)((rowlen >> 16) | (rows << 16)); // dim0[31:16] | tensor_dim1[15:0]
  g1[3] = (int)(rowlen << 16);            // tensor_dim1[31:16]=0 | tile_dim0
  g1[4] = (int)rows;                      // tile_dim1 | tile_dim2=0
  g1[5] = (int)rowlen;                    // tensor_dim0_stride[31:0]
  g1[6] = 0;                              // dim0_stride[47:32] | dim1_stride[15:0]
  g1[7] = 0;                              // dim1_stride[47:16]
  i32x4 g2 = {0, 0, 0, 0};
  i32x4 g3 = {0, 0, 0, 0};
#if defined(__clang_major__) && (__clang_major__ >= 23)
  i32x8 g4 = {0, 0, 0, 0, 0, 0, 0, 0};
  __builtin_amdgcn_tensor_load_to_lds(g0, g1, g2, g3, g4, 0);
#else
  __builtin_amdgcn_tensor_load_to_lds(g0, g1, g2, g3, 0);
#endif
}
#endif

// ---------------- fp32 -> bf16 convert ----------------
__global__ void cvt_bf16(const float* __restrict__ src, __bf16* __restrict__ dst, int n) {
  int i = blockIdx.x * blockDim.x + threadIdx.x;
  if (i < n) dst[i] = f2bf(src[i]);
}

// ============================================================
// Block-tiled WMMA GEMM for wide N: C = A(MxK) * W(NxK)^T
// Block = 128 threads (4 waves) -> 32M x 256N tile.
// A strip (32 x K bf16, K=1024 -> 64KB) staged in LDS via TDM,
// shared by all 4 waves; weights streamed from global/L2.
// MODE 0: bf16 out row-major; MODE 2: fp32 out + bias.
// ============================================================
template <int MODE>
__global__ __launch_bounds__(128) void gemm_blk(
    const __bf16* __restrict__ A, const __bf16* __restrict__ W,
    const float* __restrict__ bias, void* __restrict__ out,
    int M, int N, int K) {
  __shared__ __align__(16) __bf16 ldsA[32 * WIDTH];   // K fixed at 1024

  const int lane = threadIdx.x & 31;
  const int widx = threadIdx.x >> 5;
  const int tilesN = N >> 8;
  const int bm = blockIdx.x / tilesN;
  const int bn = blockIdx.x % tilesN;

  const __bf16* gA = A + (size_t)bm * 32 * K;
#if HAVE_TDM
  if (widx == 0) {
    tdm_load_strip(gA, 32u, (unsigned)K);
    __builtin_amdgcn_s_wait_tensorcnt(0);
  }
#else
  for (int i = threadIdx.x; i < 32 * K / 8; i += 128)
    ((unsigned long long*)ldsA)[i] = ((const unsigned long long*)gA)[i];
#endif
  __syncthreads();

  const int l16 = lane & 15;
  const int hi  = lane >> 4;

  const __bf16* wp = W + (size_t)(bn * 256 + widx * 64 + l16) * K + hi * 16;
  const __bf16* a0 = ldsA + (size_t)l16 * K;          // rows 0..15 of strip
  const __bf16* a1 = ldsA + (size_t)(16 + l16) * K;   // rows 16..31

  v8f acc[8] = {v8f{}, v8f{}, v8f{}, v8f{}, v8f{}, v8f{}, v8f{}, v8f{}};
  for (int k0 = 0; k0 < K; k0 += 32) {
    __builtin_prefetch(wp + k0 + 256, 0, 1);
    v16bf aA = load_a16(a0 + k0, hi);
    v16bf aB = load_a16(a1 + k0, hi);
    v16bf b0 = *(const v16bf*)(wp + k0);
    v16bf b1 = *(const v16bf*)(wp + (size_t)16 * K + k0);
    v16bf b2 = *(const v16bf*)(wp + (size_t)32 * K + k0);
    v16bf b3 = *(const v16bf*)(wp + (size_t)48 * K + k0);
    acc[0] = wmma_bf16(aA, b0, acc[0]);
    acc[1] = wmma_bf16(aA, b1, acc[1]);
    acc[2] = wmma_bf16(aA, b2, acc[2]);
    acc[3] = wmma_bf16(aA, b3, acc[3]);
    acc[4] = wmma_bf16(aB, b0, acc[4]);
    acc[5] = wmma_bf16(aB, b1, acc[5]);
    acc[6] = wmma_bf16(aB, b2, acc[6]);
    acc[7] = wmma_bf16(aB, b3, acc[7]);
  }

  const int ncol = bn * 256 + widx * 64 + l16;
#pragma unroll
  for (int half = 0; half < 2; ++half) {
    const int mrow = bm * 32 + half * 16 + hi * 8;  // C layout row
    if (MODE == 2) {
      float* O = (float*)out;
#pragma unroll
      for (int j = 0; j < 4; ++j) {
        int n = ncol + j * 16;
        float bv = bias[n];
#pragma unroll
        for (int v = 0; v < 8; ++v)
          O[(size_t)(mrow + v) * N + n] = acc[half * 4 + j][v] + bv;
      }
    } else {
      __bf16* O = (__bf16*)out;
#pragma unroll
      for (int j = 0; j < 4; ++j) {
        int n = ncol + j * 16;
#pragma unroll
        for (int v = 0; v < 8; ++v)
          O[(size_t)(mrow + v) * N + n] = f2bf(acc[half * 4 + j][v]);
      }
    }
  }
}

// ============================================================
// Small-N WMMA GEMM (N==64, used for k/v projections).
// One wave -> 16M x 64N tile.
// MODE 0: bf16 row-major; MODE 1: bf16 transposed out[b][n][t].
// ============================================================
template <int MODE>
__global__ __launch_bounds__(128) void gemm_small(
    const __bf16* __restrict__ A, const __bf16* __restrict__ W,
    void* __restrict__ out, int M, int N, int K, int seq) {
  const int lane = threadIdx.x & 31;
  const int widx = threadIdx.x >> 5;
  const int w    = blockIdx.x * 4 + widx;
  const int tm   = w;                      // N==64 -> one n-tile
  if (tm * 16 >= M) return;

  const int l16 = lane & 15;
  const int hi  = lane >> 4;

  const __bf16* ap = A + (size_t)(tm * 16 + l16) * K;
  const __bf16* wp = W + (size_t)l16 * K + hi * 16;

  v8f acc[4] = {v8f{}, v8f{}, v8f{}, v8f{}};
  for (int k0 = 0; k0 < K; k0 += 32) {
    __builtin_prefetch(ap + k0 + 256, 0, 1);
    v16bf a  = load_a16(ap + k0, hi);
    v16bf b0 = *(const v16bf*)(wp + k0);
    v16bf b1 = *(const v16bf*)(wp + (size_t)16 * K + k0);
    v16bf b2 = *(const v16bf*)(wp + (size_t)32 * K + k0);
    v16bf b3 = *(const v16bf*)(wp + (size_t)48 * K + k0);
    acc[0] = wmma_bf16(a, b0, acc[0]);
    acc[1] = wmma_bf16(a, b1, acc[1]);
    acc[2] = wmma_bf16(a, b2, acc[2]);
    acc[3] = wmma_bf16(a, b3, acc[3]);
  }

  const int mrow = tm * 16 + hi * 8;
  __bf16* O = (__bf16*)out;
#pragma unroll
  for (int j = 0; j < 4; ++j) {
    int n = j * 16 + l16;
#pragma unroll
    for (int v = 0; v < 8; ++v) {
      int m = mrow + v;
      if (MODE == 0) {
        O[(size_t)m * N + n] = f2bf(acc[j][v]);
      } else {  // transposed per batch: out[b][n][t]
        int bb = m / seq, t = m % seq;
        O[((size_t)bb * N + n) * seq + t] = f2bf(acc[j][v]);
      }
    }
  }
}

// ---------------- local-window MQA flash attention ----------------
// qb: [B][T][WIDTH] bf16 (feature = head*64+h), kb: [B][T][64], vT: [B][64][T]
__global__ __launch_bounds__(128) void attn_kernel(
    const __bf16* __restrict__ qb, const __bf16* __restrict__ kb,
    const __bf16* __restrict__ vT, __bf16* __restrict__ att) {
  __shared__ __align__(16) __bf16 plds[4][16 * 40];

  const int lane = threadIdx.x & 31;
  const int widx = threadIdx.x >> 5;
  const int w    = blockIdx.x * 4 + widx;
  const int tq   = SEQ / 16;
  const int b    = w / (NUM_HEADS * tq);
  const int head = (w / tq) % NUM_HEADS;
  const int t0   = (w % tq) * 16;

  const int l16 = lane & 15;
  const int hi  = lane >> 4;

  const __bf16* qp = qb + ((size_t)(b * SEQ + t0 + l16)) * WIDTH + head * HEAD_DIM;
  v16bf qa0 = load_a16(qp, hi);
  v16bf qa1 = load_a16(qp + 32, hi);

  float rmax[8], rsum[8];
#pragma unroll
  for (int v = 0; v < 8; ++v) { rmax[v] = -3.0e38f; rsum[v] = 0.f; }
  v8f o[4] = {v8f{}, v8f{}, v8f{}, v8f{}};

  int s_begin = t0 - WINDOW; if (s_begin < 0) s_begin = 0;
  s_begin &= ~31;
  int s_end = t0 + 16 + WINDOW; if (s_end > SEQ) s_end = SEQ;

  const __bf16* kbase = kb + (size_t)b * SEQ * HEAD_DIM;
  const __bf16* vbase = vT + (size_t)b * HEAD_DIM * SEQ;
  __bf16* pl = &plds[widx][0];
  const float LOG2E = 1.4426950408889634f;

  for (int sc = s_begin; sc < s_end; sc += 32) {
    // ---- logits S = Q * K^T (two 16-key subtiles x two K=32 hdim chunks) ----
    int kr0 = sc + l16;      if (kr0 >= SEQ) kr0 = SEQ - 1;
    int kr1 = sc + 16 + l16; if (kr1 >= SEQ) kr1 = SEQ - 1;
    const __bf16* kp0 = kbase + (size_t)kr0 * HEAD_DIM + hi * 16;
    const __bf16* kp1 = kbase + (size_t)kr1 * HEAD_DIM + hi * 16;
    v16bf kf00 = *(const v16bf*)(kp0);
    v16bf kf01 = *(const v16bf*)(kp0 + 32);
    v16bf kf10 = *(const v16bf*)(kp1);
    v16bf kf11 = *(const v16bf*)(kp1 + 32);
    v8f s0 = {}, s1 = {};
    s0 = wmma_bf16(qa0, kf00, s0);
    s0 = wmma_bf16(qa1, kf01, s0);
    s1 = wmma_bf16(qa0, kf10, s1);
    s1 = wmma_bf16(qa1, kf11, s1);

    // ---- mask + online softmax ----
    const int sn0 = sc + l16, sn1 = sc + 16 + l16;
    float p0[8], p1[8], scl[8];
#pragma unroll
    for (int v = 0; v < 8; ++v) {
      int trow = t0 + v + hi * 8;
      int d0 = trow - sn0, d1 = trow - sn1;
      float e0 = (sn0 < SEQ && d0 <= WINDOW && d0 >= -WINDOW) ? s0[v] * 0.125f : -1.0e30f;
      float e1 = (sn1 < SEQ && d1 <= WINDOW && d1 >= -WINDOW) ? s1[v] * 0.125f : -1.0e30f;
      float mx = fmaxf(e0, e1);
#pragma unroll
      for (int d = 1; d < 16; d <<= 1) mx = fmaxf(mx, __shfl_xor(mx, d, 32));
      float nm = fmaxf(rmax[v], mx);
      float sc_ = exp2f((rmax[v] - nm) * LOG2E);
      rmax[v] = nm;
      float q0 = exp2f((e0 - nm) * LOG2E);
      float q1 = exp2f((e1 - nm) * LOG2E);
      float ps = q0 + q1;
#pragma unroll
      for (int d = 1; d < 16; d <<= 1) ps += __shfl_xor(ps, d, 32);
      rsum[v] = rsum[v] * sc_ + ps;
      scl[v] = sc_;
      p0[v] = q0; p1[v] = q1;
    }

    // ---- rescale O, stage P through LDS (C layout -> A layout) ----
#pragma unroll
    for (int v = 0; v < 8; ++v) {
      int m = v + hi * 8;
      pl[m * 40 + l16]      = f2bf(p0[v]);
      pl[m * 40 + 16 + l16] = f2bf(p1[v]);
      float sc_ = scl[v];
      o[0][v] *= sc_; o[1][v] *= sc_; o[2][v] *= sc_; o[3][v] *= sc_;
    }
    v16bf pa = load_a16(pl + l16 * 40, hi);

    // ---- O += P * V (vT layout makes B-fragments contiguous) ----
    int sb = sc + hi * 16; if (sb > SEQ - 16) sb = SEQ - 16;
#pragma unroll
    for (int j = 0; j < 4; ++j) {
      const __bf16* vp = vbase + (size_t)(j * 16 + l16) * SEQ + sb;
      v16bf vf = *(const v16bf*)vp;
      o[j] = wmma_bf16(pa, vf, o[j]);
    }
  }

  // ---- normalize + store ----
#pragma unroll
  for (int v = 0; v < 8; ++v) {
    float inv = 1.f / rsum[v];
    int t = t0 + v + hi * 8;
    size_t base = ((size_t)(b * SEQ + t)) * WIDTH + head * HEAD_DIM;
    att[base +      l16] = f2bf(o[0][v] * inv);
    att[base + 16 + l16] = f2bf(o[1][v] * inv);
    att[base + 32 + l16] = f2bf(o[2][v] * inv);
    att[base + 48 + l16] = f2bf(o[3][v] * inv);
  }
}

// ---------------- launcher ----------------
extern "C" void kernel_launch(void* const* d_in, const int* in_sizes, int n_in,
                              void* d_out, int out_size, void* d_ws, size_t ws_size,
                              hipStream_t stream) {
  const float* x  = (const float*)d_in[0];
  const float* Wq = (const float*)d_in[1];
  const float* Wk = (const float*)d_in[2];
  const float* Wv = (const float*)d_in[3];
  const float* Wf = (const float*)d_in[4];
  const float* bf = (const float*)d_in[5];

  char* ws = (char*)d_ws;
  size_t off = 0;
  auto carve = [&](size_t bytes) -> void* {
    void* p = ws + off;
    off += (bytes + 255) & ~(size_t)255;
    return p;
  };
  __bf16* xb   = (__bf16*)carve((size_t)BT * WIDTH * 2);
  __bf16* Wqb  = (__bf16*)carve((size_t)WIDTH * WIDTH * 2);
  __bf16* Wkb  = (__bf16*)carve((size_t)HEAD_DIM * WIDTH * 2);
  __bf16* Wvb  = (__bf16*)carve((size_t)HEAD_DIM * WIDTH * 2);
  __bf16* Wfb  = (__bf16*)carve((size_t)WIDTH * WIDTH * 2);
  __bf16* qb   = (__bf16*)carve((size_t)BT * WIDTH * 2);
  __bf16* kbuf = (__bf16*)carve((size_t)BT * HEAD_DIM * 2);
  __bf16* vT   = (__bf16*)carve((size_t)BT * HEAD_DIM * 2);
  __bf16* att  = (__bf16*)carve((size_t)BT * WIDTH * 2);

  auto cvt = [&](const float* s, __bf16* d, int n) {
    cvt_bf16<<<(n + 255) / 256, 256, 0, stream>>>(s, d, n);
  };
  cvt(x,  xb,  BT * WIDTH);
  cvt(Wq, Wqb, WIDTH * WIDTH);
  cvt(Wk, Wkb, HEAD_DIM * WIDTH);
  cvt(Wv, Wvb, HEAD_DIM * WIDTH);
  cvt(Wf, Wfb, WIDTH * WIDTH);

  // q = x @ Wq^T  -> bf16 [BT][1024]   (block-tiled, TDM-staged A)
  gemm_blk<0><<<(BT / 32) * (WIDTH / 256), 128, 0, stream>>>(
      xb, Wqb, nullptr, qb, BT, WIDTH, WIDTH);
  // k = x @ Wk^T  -> bf16 [BT][64]
  gemm_small<0><<<(BT / 16) / 4, 128, 0, stream>>>(
      xb, Wkb, kbuf, BT, HEAD_DIM, WIDTH, SEQ);
  // v = x @ Wv^T  -> bf16 transposed [B][64][T]
  gemm_small<1><<<(BT / 16) / 4, 128, 0, stream>>>(
      xb, Wvb, vT, BT, HEAD_DIM, WIDTH, SEQ);
  // local-window attention -> bf16 [BT][1024]
  attn_kernel<<<(BATCH * NUM_HEADS * (SEQ / 16)) / 4, 128, 0, stream>>>(
      qb, kbuf, vT, att);
  // y = att @ Wf^T + bf -> fp32 d_out  (block-tiled, TDM-staged A)
  gemm_blk<2><<<(BT / 32) * (WIDTH / 256), 128, 0, stream>>>(
      att, Wfb, bf, d_out, BT, WIDTH, WIDTH);
}
